// TemporalAttention_64553358459243
// MI455X (gfx1250) — matrix operand order repaired and checked by reading, not compiled, not run on hardware
//
#include <hip/hip_runtime.h>
#include <hip/hip_bf16.h>
#include <math.h>

#define MDIM 512
#define PDIM 256
#define BDIM 128
#define TLEN 1024
#define TROWS 64
#define LDSROW 520   // 512 + 8 bf16 pad -> kills LDS bank conflicts, keeps 16B align

typedef __attribute__((ext_vector_type(16))) __bf16 v16bf;
typedef __attribute__((ext_vector_type(8)))  float  v8f;

union Frag16 { uint4 u[2]; v16bf v; };

__device__ __forceinline__ unsigned short f2bf(float f) {
    union { float f; unsigned u; } x; x.f = f;
    unsigned r = x.u + 0x7FFFu + ((x.u >> 16) & 1u);   // round-to-nearest-even
    return (unsigned short)(r >> 16);
}

// ---------------------------------------------------------------------------
// Pack U_d (stored [n][m] row-major) into per-lane WMMA B-fragment order:
// fragment f = s*32 + jg  (s = K-step 0..15, jg = N-tile 0..31)
// element e = (f*512) + lane*16 + h, holding B[k][n] with
//   n = jg*16 + (lane&15),  k = s*32 + (lane<16 ? 0 : 16) + h
// ---------------------------------------------------------------------------
__global__ void k_prep_upack(const float* __restrict__ U, unsigned short* __restrict__ Up) {
    int e    = blockIdx.x * 256 + threadIdx.x;   // 0 .. 262143
    int h    = e & 15;
    int lane = (e >> 4) & 31;
    int f    = e >> 9;
    int s    = f >> 5;
    int jg   = f & 31;
    int n    = jg * 16 + (lane & 15);
    int k    = s * 32 + ((lane < 16) ? 0 : 16) + h;
    Up[e] = f2bf(U[(size_t)n * MDIM + k]);
}

// wq[b][n] = sum_c query[b][c] * W_d[n][c],  query = [d_t | s_t]
__global__ void k_prep_wq(const float* __restrict__ dt, const float* __restrict__ st,
                          const float* __restrict__ W, float* __restrict__ wq) {
    __shared__ float q[2 * PDIM];
    int b = blockIdx.x, tid = threadIdx.x;
    for (int i = tid; i < 2 * PDIM; i += 256)
        q[i] = (i < PDIM) ? dt[b * PDIM + i] : st[b * PDIM + (i - PDIM)];
    __syncthreads();
    const float4* q4 = (const float4*)q;
    for (int n = tid; n < MDIM; n += 256) {
        const float4* w4 = (const float4*)(W + (size_t)n * (2 * PDIM));
        float acc = 0.f;
        for (int c = 0; c < (2 * PDIM) / 4; ++c) {
            float4 a = q4[c], w = w4[c];
            acc += a.x * w.x + a.y * w.y + a.z * w.z + a.w * w.w;
        }
        wq[b * MDIM + n] = acc;
    }
}

// ---------------------------------------------------------------------------
// Main fused kernel: score[b,t] = sum_n tanh(wq[b,n] + (H[b] @ U^T)[t,n]) * v[n]
// One WG per (b, 64-row t-block). 8 waves: wave w -> row-tile w/2, n-half w&1.
// ---------------------------------------------------------------------------
__global__ void __launch_bounds__(256)
k_attn(const float* __restrict__ H, const unsigned short* __restrict__ Up,
       const float* __restrict__ wq, const float* __restrict__ vd,
       float* __restrict__ out)
{
    __shared__ unsigned short Ht[TROWS * LDSROW];     // 65 KB bf16 H tile
    __shared__ float score_s[TROWS];

    int b    = blockIdx.x >> 4;
    int t0   = (blockIdx.x & 15) * TROWS;
    int tid  = threadIdx.x;
    int lane = tid & 31;
    int w    = tid >> 5;
    int rt   = w >> 1;          // row tile 0..3
    int nh   = w & 1;           // which 256-wide n half

    if (tid < TROWS) score_s[tid] = 0.f;

    // cooperative load + f32->bf16 convert of the 64x512 H tile
    const float* Hb = H + ((size_t)b * TLEN + t0) * MDIM;
    for (int idx = tid; idx < TROWS * (MDIM / 4); idx += 256) {
        int row = idx >> 7;                 // MDIM/4 = 128
        int c4  = idx & 127;
        float4 v = ((const float4*)Hb)[row * (MDIM / 4) + c4];
        uint2 pack;
        pack.x = (unsigned)f2bf(v.x) | ((unsigned)f2bf(v.y) << 16);
        pack.y = (unsigned)f2bf(v.z) | ((unsigned)f2bf(v.w) << 16);
        *(uint2*)&Ht[row * LDSROW + c4 * 4] = pack;
    }
    __syncthreads();

    const uint4* lds4 = (const uint4*)Ht;
    const uint4* up4  = (const uint4*)Up;

    // A-fragment per-lane addressing (contiguous k = 16h + s, matching packed B)
    int rowl  = rt * 16 + (lane & 15);
    int abase = (lane < 16) ? 0 : 16;
    int ncol  = lane & 15;

    float part[8];
#pragma unroll
    for (int r = 0; r < 8; ++r) part[r] = 0.f;

    for (int jb = 0; jb < 4; ++jb) {            // 4 n-tiles accumulated at once
        v8f acc[4];
#pragma unroll
        for (int jj = 0; jj < 4; ++jj)
#pragma unroll
            for (int r = 0; r < 8; ++r) acc[jj][r] = 0.f;

#pragma unroll 4
        for (int s = 0; s < 16; ++s) {          // K = 512 in steps of 32
            Frag16 a;
            int off = rowl * LDSROW + s * 32 + abase;
            a.u[0] = lds4[off >> 3];
            a.u[1] = lds4[(off + 8) >> 3];
#pragma unroll
            for (int jj = 0; jj < 4; ++jj) {
                int jg = nh * 16 + jb * 4 + jj;
                Frag16 bm;
                int fo = ((s * 32 + jg) * 32 + lane) * 2;   // uint4 index
                bm.u[0] = up4[fo];
                bm.u[1] = up4[fo + 1];
                acc[jj] = __builtin_amdgcn_wmma_f32_16x16x32_bf16(
                    false, a.v, false, bm.v, (short)0, acc[jj], false, false);
            }
        }
        // fused epilogue: tanh(uh + wq) * v, summed over this n-tile's columns
#pragma unroll
        for (int jj = 0; jj < 4; ++jj) {
            int jg = nh * 16 + jb * 4 + jj;
            float wqv = wq[b * MDIM + jg * 16 + ncol];
            float vv  = vd[jg * 16 + ncol];
#pragma unroll
            for (int r = 0; r < 8; ++r)
                part[r] += tanhf(acc[jj][r] + wqv) * vv;
        }
    }

    // reduce over the 16 N-lanes of each half-wave
#pragma unroll
    for (int m = 1; m < 16; m <<= 1)
#pragma unroll
        for (int r = 0; r < 8; ++r)
            part[r] += __shfl_xor(part[r], m, 32);

    if ((lane & 15) == 0) {
        int roff = (lane < 16) ? 0 : 8;         // C layout: VGPR r -> M=r / M=8+r
#pragma unroll
        for (int r = 0; r < 8; ++r)
            atomicAdd(&score_s[rt * 16 + roff + r], part[r]);
    }
    __syncthreads();
    if (tid < TROWS) out[(size_t)b * TLEN + t0 + tid] = score_s[tid];
}

// in-place softmax over T=1024 per batch row
__global__ void k_softmax(float* __restrict__ out) {
    __shared__ float red[256];
    int b = blockIdx.x, tid = threadIdx.x;
    float* row = out + (size_t)b * TLEN;
    float v[4];
    float lmax = -1e30f;
#pragma unroll
    for (int k = 0; k < 4; ++k) { v[k] = row[tid + k * 256]; lmax = fmaxf(lmax, v[k]); }
    red[tid] = lmax; __syncthreads();
    for (int s = 128; s > 0; s >>= 1) { if (tid < s) red[tid] = fmaxf(red[tid], red[tid + s]); __syncthreads(); }
    float mx = red[0]; __syncthreads();
    float lsum = 0.f;
#pragma unroll
    for (int k = 0; k < 4; ++k) { v[k] = expf(v[k] - mx); lsum += v[k]; }
    red[tid] = lsum; __syncthreads();
    for (int s = 128; s > 0; s >>= 1) { if (tid < s) red[tid] += red[tid + s]; __syncthreads(); }
    float inv = 1.f / red[0];
#pragma unroll
    for (int k = 0; k < 4; ++k) row[tid + k * 256] = v[k] * inv;
}

extern "C" void kernel_launch(void* const* d_in, const int* in_sizes, int n_in,
                              void* d_out, int out_size, void* d_ws, size_t ws_size,
                              hipStream_t stream) {
    const float* d_t = (const float*)d_in[0];
    const float* s_t = (const float*)d_in[1];
    const float* H   = (const float*)d_in[2];
    /* d_in[3] = T (scalar, fixed 1024) */
    const float* W_d = (const float*)d_in[4];
    const float* U_d = (const float*)d_in[5];
    const float* v_d = (const float*)d_in[6];
    float* out = (float*)d_out;

    unsigned short* Up = (unsigned short*)d_ws;                        // 512 KB packed bf16 U^T
    float* wq = (float*)((char*)d_ws + (size_t)MDIM * MDIM * sizeof(unsigned short)); // 256 KB

    k_prep_upack<<<(MDIM * MDIM) / 256, 256, 0, stream>>>(U_d, Up);
    k_prep_wq<<<BDIM, 256, 0, stream>>>(d_t, s_t, W_d, wq);
    k_attn<<<BDIM * (TLEN / TROWS), 256, 0, stream>>>(H, Up, wq, v_d, out);
    k_softmax<<<BDIM, 256, 0, stream>>>(out);
}
